// CoreModule_16587163697950
// MI455X (gfx1250) — matrix operand order repaired and checked by reading, not compiled, not run on hardware
//
#include <hip/hip_runtime.h>
#include <hip/hip_bf16.h>

#define N 4096
#define D 256
#define H 4
#define TILE_I 16
#define NT (N / 32)          // 128 j-tiles

typedef __attribute__((ext_vector_type(16))) __bf16 v16bf;
typedef __attribute__((ext_vector_type(8)))  float  v8f;

__device__ __forceinline__ float elu_f(float x) {
    return x > 0.f ? x : (__expf(x) - 1.f);
}

__device__ __forceinline__ unsigned short f2bf(float f) {
    unsigned u = __builtin_bit_cast(unsigned, f);
    unsigned r = u + 0x7FFFu + ((u >> 16) & 1u);   // round-to-nearest-even
    return (unsigned short)(r >> 16);
}

// ---------------------------------------------------------------------------
// Kernel 0: h = x*conv_w + conv_b ; store hT (bf16, [d][j] transposed layout
// for contiguous WMMA A-operand loads) ; Wh1[h][i] = h_i . a1[h],
// Wh2[h][i] = h_i . a2[h].
// ---------------------------------------------------------------------------
__global__ void gat_prep_kernel(const float* __restrict__ x,
                                const float* __restrict__ conv_w,
                                const float* __restrict__ conv_b,
                                const float* __restrict__ a,
                                unsigned short* __restrict__ hT,
                                float* __restrict__ Wh1,
                                float* __restrict__ Wh2) {
    __shared__ float sa[H * 2 * D];      // 8 KB: a[h][0..2D)
    const int t = threadIdx.x;
    for (int k = t; k < H * 2 * D; k += blockDim.x) sa[k] = a[k];
    __syncthreads();

    const float cw = conv_w[0], cb = conv_b[0];
    const int i = blockIdx.x * blockDim.x + t;       // row index

    float d1[H], d2[H];
#pragma unroll
    for (int hh = 0; hh < H; ++hh) { d1[hh] = 0.f; d2[hh] = 0.f; }

    for (int d = 0; d < D; ++d) {
        float hv = x[(size_t)i * D + d] * cw + cb;
        hT[(size_t)d * N + i] = f2bf(hv);            // coalesced across threads
#pragma unroll
        for (int hh = 0; hh < H; ++hh) {
            d1[hh] += hv * sa[hh * 2 * D + d];
            d2[hh] += hv * sa[hh * 2 * D + D + d];
        }
    }
#pragma unroll
    for (int hh = 0; hh < H; ++hh) {
        Wh1[hh * N + i] = d1[hh];
        Wh2[hh * N + i] = d2[hh];
    }
}

// ---------------------------------------------------------------------------
// Kernel 1: per (head, row i) masked-softmax stats: row max m and 1/Z.
// One wave32 per (head,row), online max+sum, butterfly combine.
// ---------------------------------------------------------------------------
__global__ void gat_stats_kernel(const int*   __restrict__ adj,
                                 const float* __restrict__ Wh1,
                                 const float* __restrict__ Wh2,
                                 float* __restrict__ Mbuf,
                                 float* __restrict__ Zbuf) {
    const int wid  = (blockIdx.x * blockDim.x + threadIdx.x) >> 5;  // [0, H*N)
    const int lane = threadIdx.x & 31;
    const int head = wid / N;
    const int i    = wid % N;

    const float s = Wh1[head * N + i];
    const int*   arow = adj + (size_t)i * N;
    const float* trow = Wh2 + head * N;

    float m = -3.0e38f, sum = 0.f;
    for (int j = lane; j < N; j += 32) {             // coalesced
        float l = (arow[j] > 0) ? elu_f(s + trow[j]) : -9.0e15f;
        if (l > m) { sum = sum * __expf(m - l) + 1.f; m = l; }
        else       { sum += __expf(l - m); }
    }
#pragma unroll
    for (int off = 16; off > 0; off >>= 1) {
        float mo = __shfl_xor(m,   off, 32);
        float so = __shfl_xor(sum, off, 32);
        float mn = fmaxf(m, mo);
        sum = sum * __expf(m - mn) + so * __expf(mo - mn);
        m = mn;
    }
    if (lane == 0) {
        Mbuf[head * N + i] = m;
        Zbuf[head * N + i] = 1.f / sum;
    }
}

// ---------------------------------------------------------------------------
// Kernel 2: fused attention matmul + epilogue.
// mean_h(att_h @ h) = (mean_h att_h) @ h -> one bf16 matmul.
// Block = 16 rows; 4 waves partition the 256 d-columns (4 x 64), so each
// wave carries only 4 accumulator tiles (32 VGPRs). The shared B operand
// (head-summed softmax weights) is computed cooperatively once per block
// and staged through a double-buffered LDS tile, software-pipelined one
// j-tile ahead; one __syncthreads per iteration sits between A-register
// loads and the WMMAs so load latency hides under the exp/elu VALU burst.
// ---------------------------------------------------------------------------
__global__ __launch_bounds__(128) void gat_wmma_kernel(
        const int*   __restrict__ adj,
        const float* __restrict__ x,
        const float* __restrict__ conv_w,
        const float* __restrict__ conv_b,
        const unsigned short* __restrict__ hT,
        const float* __restrict__ Wh1,
        const float* __restrict__ Wh2,
        const float* __restrict__ Mbuf,
        const float* __restrict__ Zbuf,
        const float* __restrict__ bias,
        float* __restrict__ out) {
    __shared__ float lds_out[D * TILE_I];                 // [d][i], 16 KB
    __shared__ unsigned short lds_B[2][TILE_I * 32];      // weights, 2x1 KB
    __shared__ float lds_ss[TILE_I * 8];
    __shared__ float lds_nrm[TILE_I];

    const int t    = threadIdx.x;
    const int lane = t & 31;
    const int wave = __builtin_amdgcn_readfirstlane(t >> 5);
    const int i0   = blockIdx.x * TILE_I;
    const int il   = lane & 15;                      // B column / output row
    const int hi   = lane >> 4;                      // half-wave select
    const int kb   = hi * 8;                         // B operand K base
    const int ab   = hi * 8;                         // A operand K base
    const int cg0  = wave * 4;                       // this wave's d-tiles

    // weight-compute mapping: thread -> (row iw, 4 consecutive k at kw)
    const int iw = t >> 3;                           // 0..15
    const int kw = (t & 7) * 4;                      // 0..28

    // per-thread, per-head softmax row constants (for row iw)
    float sH[H], mH[H], izH[H];
#pragma unroll
    for (int h = 0; h < H; ++h) {
        sH[h]  = Wh1[h * N + i0 + iw];
        mH[h]  = Mbuf[h * N + i0 + iw];
        izH[h] = Zbuf[h * N + i0 + iw];
    }
    const int* arow = adj + (size_t)(i0 + iw) * N;

    // head-summed masked softmax weights: 4 per thread -> lds_B[jt&1]
    auto computeB = [&](int jt) {
        const int j0 = jt * 32;
        const int4 am = *(const int4*)(arow + j0 + kw);
        float w0 = 0.f, w1 = 0.f, w2 = 0.f, w3 = 0.f;
#pragma unroll
        for (int h = 0; h < H; ++h) {
            const float4 tv = *(const float4*)(Wh2 + h * N + j0 + kw);
            const float s = sH[h], mi = mH[h], iz = izH[h];
            w0 += __expf(elu_f(s + tv.x) - mi) * iz;
            w1 += __expf(elu_f(s + tv.y) - mi) * iz;
            w2 += __expf(elu_f(s + tv.z) - mi) * iz;
            w3 += __expf(elu_f(s + tv.w) - mi) * iz;
        }
        unsigned b0 = (am.x > 0) ? f2bf(w0) : 0u;
        unsigned b1 = (am.y > 0) ? f2bf(w1) : 0u;
        unsigned b2 = (am.z > 0) ? f2bf(w2) : 0u;
        unsigned b3 = (am.w > 0) ? f2bf(w3) : 0u;
        uint2 packed;
        packed.x = b0 | (b1 << 16);
        packed.y = b2 | (b3 << 16);
        *(uint2*)&lds_B[jt & 1][iw * 32 + kw] = packed;
    };

    v8f acc[4];
#pragma unroll
    for (int c = 0; c < 4; ++c) acc[c] = {};

    union Au { v16bf v; uint4 q[2]; };

    computeB(0);
    __syncthreads();

    for (int jt = 0; jt < NT; ++jt) {
        const int j0 = jt * 32;

        // A operand registers for this wave's 4 d-tiles (issued early;
        // per-lane halves 0..7 = K ab..ab+7, halves 8..15 = K ab+16..ab+23)
        Au aq[4];
#pragma unroll
        for (int c = 0; c < 4; ++c) {
            const uint4* hp = reinterpret_cast<const uint4*>(
                hT + (size_t)((cg0 + c) * 16 + il) * N + j0 + ab);
            aq[c].q[0] = hp[0];
            aq[c].q[1] = hp[2];
        }

        // pipeline: next tile's weights while A loads are in flight
        if (jt + 1 < NT) computeB(jt + 1);

        // B operand from LDS (stores barrier-separated at previous iter)
        union { v16bf v; uint4 q[2]; } B;
        const uint4* bq = reinterpret_cast<const uint4*>(
            &lds_B[jt & 1][il * 32 + kb]);
        B.q[0] = bq[0];
        B.q[1] = bq[2];

        __syncthreads();   // publishes B(jt+1); A/B regs already local

#pragma unroll
        for (int c = 0; c < 4; ++c) {
            acc[c] = __builtin_amdgcn_wmma_f32_16x16x32_bf16(
                false, aq[c].v, false, B.v, (short)0, acc[c], false, false);
        }
    }

    // waves own disjoint d columns: direct stores, no reduction needed
#pragma unroll
    for (int c = 0; c < 4; ++c)
#pragma unroll
        for (int r = 0; r < 8; ++r) {
            int d = (cg0 + c) * 16 + r + hi * 8;     // C layout: M = r (+8 hi)
            lds_out[d * TILE_I + il] = acc[c][r];
        }
    __syncthreads();

    // ---- epilogue: mean over heads, mix, elu, row L2 norm, +bias
    const float cw = conv_w[0], cb = conv_b[0];
    const int row = t >> 3;                          // 0..15
    const int d0  = (t & 7) * 32;                    // 32 d's per thread
    float ssq = 0.f;
    for (int d = d0; d < d0 + 32; ++d) {
        float hp = lds_out[d * TILE_I + row] * (1.f / H);
        float hv = x[(size_t)(i0 + row) * D + d] * cw + cb;
        float o  = elu_f(0.5f * hp + 0.5f * hv);
        lds_out[d * TILE_I + row] = o;
        ssq += o * o;
    }
    lds_ss[row * 8 + (t & 7)] = ssq;
    __syncthreads();
    if (t < TILE_I) {
        float v = 0.f;
#pragma unroll
        for (int k = 0; k < 8; ++k) v += lds_ss[t * 8 + k];
        lds_nrm[t] = fmaxf(sqrtf(v), 1e-12f);
    }
    __syncthreads();
    const float inv_n = 1.f / lds_nrm[row];
    for (int d = d0; d < d0 + 32; ++d) {
        out[(size_t)(i0 + row) * D + d] =
            lds_out[d * TILE_I + row] * inv_n + bias[d];
    }
}

// ---------------------------------------------------------------------------
extern "C" void kernel_launch(void* const* d_in, const int* in_sizes, int n_in,
                              void* d_out, int out_size, void* d_ws, size_t ws_size,
                              hipStream_t stream) {
    const float* x      = (const float*)d_in[0];
    const int*   adj    = (const int*)  d_in[1];
    const float* conv_w = (const float*)d_in[2];
    const float* conv_b = (const float*)d_in[3];
    const float* a      = (const float*)d_in[4];
    const float* bias   = (const float*)d_in[5];
    float* out = (float*)d_out;

    // workspace layout (~2.25 MB)
    unsigned short* hT = (unsigned short*)d_ws;                  // D*N*2 = 2 MB
    float* Wh1  = (float*)((char*)d_ws + (size_t)D * N * 2);     // H*N
    float* Wh2  = Wh1 + H * N;
    float* Mbuf = Wh2 + H * N;
    float* Zbuf = Mbuf + H * N;

    gat_prep_kernel<<<N / 256, 256, 0, stream>>>(x, conv_w, conv_b, a, hT, Wh1, Wh2);
    gat_stats_kernel<<<(H * N * 32) / 256, 256, 0, stream>>>(adj, Wh1, Wh2, Mbuf, Zbuf);
    gat_wmma_kernel<<<N / TILE_I, 128, 0, stream>>>(adj, x, conv_w, conv_b, hT,
                                                    Wh1, Wh2, Mbuf, Zbuf, bias, out);
}
